// GraphGrowingDecoder_56942676410889
// MI455X (gfx1250) — compile-verified, matrix operand order checked by reading, not run on hardware
//
#include <hip/hip_runtime.h>

typedef unsigned short u16;
typedef unsigned int   u32;

// ---------------- problem constants ----------------
#define NSTEP   9        // K_LITS + 1
#define KLITS   8
#define BB      64
#define HH      256
#define MAXMEM  2304     // 2 * N_MEM / B
#define MAXSYM  256      // 2 * N_SY / B
#define NCL 4096
#define NLI 16384
#define NSY 8192
#define NTE 32768
#define NVA 8192

// output (float) offsets, concatenated flat in reference return order
#define O_CONT  0
#define O_POL   576
#define O_PS    1152
#define O_NA    148608
#define O_AT    151488
#define O_AV    158400
#define O_AF    204480
#define O_TNL   794304
#define O_TPOL  794368
#define O_TPIDX 794880
#define O_TNARG 795392
#define O_TAT   795904
#define O_TAV   797952
#define O_TAF   800000

// ---------------- WMMA types ----------------
typedef __bf16 bf16_t;
typedef bf16_t v16bf __attribute__((ext_vector_type(16)));
typedef float  v8f   __attribute__((ext_vector_type(8)));

struct __attribute__((aligned(16))) Q16 { u32 x, y, z, w; };
union FragU { struct { Q16 lo, hi; } q; v16bf v; };

__device__ __forceinline__ u16 f2bf(float f) {
  u32 u = __float_as_uint(f);
  u32 r = u + 0x7FFFu + ((u >> 16) & 1u);
  return (u16)(r >> 16);
}
__device__ __forceinline__ float bf2f(u16 h) { return __uint_as_float(((u32)h) << 16); }
// packed pair helpers: u32 holds bf16 elems {lo = dim 2k, hi = dim 2k+1}
__device__ __forceinline__ float bfp_lo(u32 u) { return __uint_as_float(u << 16); }
__device__ __forceinline__ float bfp_hi(u32 u) { return __uint_as_float(u & 0xFFFF0000u); }

// A-fragment (16x32 bf16, MxK) per ISA 7.12.2: lane m = lane&15, half = lane>>4
// elems 0..7  -> K = kc + half*8 + e ; elems 8..15 -> K = kc + 16 + half*8 + (e-8)
// B-fragment: symmetric layout with n = lane&15 over W^T (W stored NxK row-major).
__device__ __forceinline__ v16bf load_frag(const u16* base, int ld, int row0, int lane, int kc) {
  const u16* p = base + (size_t)(row0 + (lane & 15)) * (size_t)ld + kc + ((lane >> 4) << 3);
  FragU u;
  u.q.lo = *(const Q16*)(p);
  u.q.hi = *(const Q16*)(p + 16);
  return u.v;
}

// ---------------- generic bf16 WMMA GEMM: C = A(MxK) @ W(NxK)^T + bias ----------------
// tile id is N-fastest so the 8 waves of a block share the same A fragments (L0/L2 hits).
template <bool RELU, bool OUTBF>
__global__ __launch_bounds__(256) void gemm_bf16_kernel(
    const u16* __restrict__ A, int lda,
    const u16* __restrict__ W,            // N x K row-major (acts as B^T)
    const float* __restrict__ bias,
    void* __restrict__ C, int ldc,
    int tilesM, int tilesN, int K) {
  int wave = threadIdx.x >> 5;
  int lane = threadIdx.x & 31;
  int tile = blockIdx.x * 8 + wave;
  if (tile >= tilesM * tilesN) return;           // wave-uniform: EXEC stays all-ones
  int tn = tile % tilesN, tm = tile / tilesN;
  int row0 = tm << 4, col0 = tn << 4;
  v8f acc = {0.f, 0.f, 0.f, 0.f, 0.f, 0.f, 0.f, 0.f};
  for (int kc = 0; kc < K; kc += 32) {
    v16bf a = load_frag(A, lda, row0, lane, kc);
    v16bf b = load_frag(W, K, col0, lane, kc);
    acc = __builtin_amdgcn_wmma_f32_16x16x32_bf16(false, a, false, b, (short)0, acc, false, false);
  }
  int n = col0 + (lane & 15);
  int mbase = row0 + ((lane >> 4) << 3);
  float bs = bias[n];
  for (int r = 0; r < 8; ++r) {
    float v = acc[r] + bs;
    if (RELU) v = fmaxf(v, 0.f);
    if (OUTBF) ((u16*)C)[(size_t)(mbase + r) * (size_t)ldc + n] = f2bf(v);
    else       ((float*)C)[(size_t)(mbase + r) * (size_t)ldc + n] = v;
  }
}

// dual-output GEMM: shares each A fragment across two WMMA chains (K and V projections),
// halving activation traffic on the dominant (B*2304 x 256 x 256) GEMM.
template <bool OUTBF>
__global__ __launch_bounds__(256) void gemm_bf16_dual_kernel(
    const u16* __restrict__ A, int lda,
    const u16* __restrict__ W1, const float* __restrict__ b1, void* __restrict__ C1,
    const u16* __restrict__ W2, const float* __restrict__ b2, void* __restrict__ C2,
    int ldc, int tilesM, int tilesN, int K) {
  int wave = threadIdx.x >> 5;
  int lane = threadIdx.x & 31;
  int tile = blockIdx.x * 8 + wave;
  if (tile >= tilesM * tilesN) return;
  int tn = tile % tilesN, tm = tile / tilesN;
  int row0 = tm << 4, col0 = tn << 4;
  v8f acc1 = {0.f, 0.f, 0.f, 0.f, 0.f, 0.f, 0.f, 0.f};
  v8f acc2 = {0.f, 0.f, 0.f, 0.f, 0.f, 0.f, 0.f, 0.f};
  for (int kc = 0; kc < K; kc += 32) {
    v16bf a  = load_frag(A, lda, row0, lane, kc);
    v16bf w1 = load_frag(W1, K, col0, lane, kc);
    v16bf w2 = load_frag(W2, K, col0, lane, kc);
    acc1 = __builtin_amdgcn_wmma_f32_16x16x32_bf16(false, a, false, w1, (short)0, acc1, false, false);
    acc2 = __builtin_amdgcn_wmma_f32_16x16x32_bf16(false, a, false, w2, (short)0, acc2, false, false);
  }
  int n = col0 + (lane & 15);
  int mbase = row0 + ((lane >> 4) << 3);
  float bs1 = b1[n], bs2 = b2[n];
  for (int r = 0; r < 8; ++r) {
    float v1 = acc1[r] + bs1;
    float v2 = acc2[r] + bs2;
    size_t idx = (size_t)(mbase + r) * (size_t)ldc + n;
    if (OUTBF) { ((u16*)C1)[idx] = f2bf(v1); ((u16*)C2)[idx] = f2bf(v2); }
    else       { ((float*)C1)[idx] = v1;     ((float*)C2)[idx] = v2; }
  }
}

// ---------------- small utility kernels ----------------
__global__ void zero_kernel(u32* p, int n) {
  int i = blockIdx.x * blockDim.x + threadIdx.x;
  if (i < n) p[i] = 0u;
}

__global__ void cvt_bf16_kernel(const float* __restrict__ in, u16* __restrict__ out, int n) {
  int i = blockIdx.x * blockDim.x + threadIdx.x;
  if (i < n) out[i] = f2bf(in[i]);
}

__global__ void vec_linear_kernel(const float* __restrict__ x, const float* __restrict__ W,
                                  const float* __restrict__ b, float* __restrict__ out,
                                  int N, int K) {
  int j = blockIdx.x * blockDim.x + threadIdx.x;
  if (j >= N) return;
  float a = b[j];
  const float* w = W + (size_t)j * K;
  for (int k = 0; k < K; ++k) a += x[k] * w[k];
  out[j] = a;
}

// per-batch segment starts via binary search on sorted batch arrays + prefix offsets
__global__ void starts_kernel(const int* s0, int n0, const int* s1, int n1, const int* s2, int n2,
                              const int* s3, int n3, const int* s4, int n4,
                              int* starts /*5*65*/, int* pre /*5*64*/) {
  const int* arr[5] = {s0, s1, s2, s3, s4};
  int ns[5] = {n0, n1, n2, n3, n4};
  int tid = threadIdx.x;
  for (int idx = tid; idx < 5 * 65; idx += blockDim.x) {
    int a = idx / 65, b = idx % 65;
    int lo = 0, hi = ns[a];
    while (lo < hi) { int mid = (lo + hi) >> 1; if (arr[a][mid] < b) lo = mid + 1; else hi = mid; }
    starts[a * 65 + b] = lo;
  }
  __syncthreads();
  for (int idx = tid; idx < 5 * 64; idx += blockDim.x) {
    int a = idx / 64, b = idx % 64;
    int p = 0;
    for (int a2 = 0; a2 < a; ++a2) p += starts[a2 * 65 + b + 1] - starts[a2 * 65 + b];
    pre[a * 64 + b] = p;
  }
}

__global__ __launch_bounds__(256) void scatter_mem_kernel(
    const float* x0, const float* x1, const float* x2, const float* x3, const float* x4,
    const int* s0, const int* s1, const int* s2, const int* s3, const int* s4,
    const int* __restrict__ starts, const int* __restrict__ pre,
    u16* __restrict__ mem_e, int* __restrict__ mem_mask) {
  int gid = blockIdx.x, tid = threadIdx.x;
  int a, i; const float* x; const int* seg;
  if      (gid < NCL)                     { a = 0; i = gid;                     x = x0; seg = s0; }
  else if (gid < NCL + NLI)               { a = 1; i = gid - NCL;               x = x1; seg = s1; }
  else if (gid < NCL + NLI + NSY)         { a = 2; i = gid - NCL - NLI;         x = x2; seg = s2; }
  else if (gid < NCL + NLI + NSY + NTE)   { a = 3; i = gid - NCL - NLI - NSY;   x = x3; seg = s3; }
  else                                    { a = 4; i = gid - NCL - NLI - NSY - NTE; x = x4; seg = s4; }
  int b = seg[i];
  int col = pre[a * 64 + b] + (i - starts[a * 65 + b]);
  if (col >= MAXMEM) return;
  mem_e[(size_t)(b * MAXMEM + col) * HH + tid] = f2bf(x[(size_t)i * HH + tid]);
  if (tid == 0) mem_mask[b * MAXMEM + col] = 1;
}

__global__ __launch_bounds__(256) void scatter_sym_kernel(
    const float* __restrict__ xs, const int* __restrict__ seg,
    const int* __restrict__ starts, u16* __restrict__ sym_e, int* __restrict__ sym_mask) {
  int i = blockIdx.x, tid = threadIdx.x;
  int b = seg[i];
  int col = i - starts[2 * 65 + b];
  if (col >= MAXSYM) return;
  sym_e[(size_t)(b * MAXSYM + col) * HH + tid] = f2bf(xs[(size_t)i * HH + tid]);
  if (tid == 0) sym_mask[b * MAXSYM + col] = 1;
}

// residual + layernorm; res_stride==0 broadcasts res (used for the constant query_embed)
__global__ __launch_bounds__(256) void ln_kernel(
    const float* __restrict__ x, const float* __restrict__ res, int res_stride,
    const float* __restrict__ g, const float* __restrict__ bt,
    float* __restrict__ outf, u16* __restrict__ outbf) {
  __shared__ float red[256];
  int row = blockIdx.x, tid = threadIdx.x;
  float v = x[row * HH + tid] + res[row * res_stride + tid];
  red[tid] = v; __syncthreads();
  for (int off = 128; off; off >>= 1) { if (tid < off) red[tid] += red[tid + off]; __syncthreads(); }
  float mean = red[0] * (1.f / HH); __syncthreads();
  float dv = v - mean;
  red[tid] = dv * dv; __syncthreads();
  for (int off = 128; off; off >>= 1) { if (tid < off) red[tid] += red[tid + off]; __syncthreads(); }
  float var = red[0] * (1.f / HH);
  float y = dv * rsqrtf(var + 1e-5f) * g[tid] + bt[tid];
  outf[row * HH + tid] = y;
  if (outbf) outbf[row * HH + tid] = f2bf(y);
}

// self-attention: single (constant) query token vs [hist_0..hist_{s-1}, query] keys
__global__ __launch_bounds__(256) void self_attn_kernel(
    const float* __restrict__ qproj, const float* __restrict__ kq, const float* __restrict__ vq,
    const float* __restrict__ Kh, const float* __restrict__ Vh,   // B x 8 x 256
    u16* __restrict__ o_bf, int s) {
  __shared__ float qs[256];
  __shared__ float sc[4][16];
  int b = blockIdx.x, tid = threadIdx.x;
  int L = s + 1;
  qs[tid] = qproj[tid];
  __syncthreads();
  if (tid < 64) {
    int h = tid >> 4, u = tid & 15;
    if (u < L) {
      const float* krow = (u < s) ? (Kh + (size_t)(b * KLITS + u) * HH) : kq;
      float acc = 0.f;
      for (int d = 0; d < 64; ++d) acc += qs[h * 64 + d] * krow[h * 64 + d];
      sc[h][u] = acc * 0.125f;
    }
  }
  __syncthreads();
  if (tid < 4) {
    int h = tid;
    float mx = -__builtin_inff();
    for (int u = 0; u < L; ++u) mx = fmaxf(mx, sc[h][u]);
    float sm = 0.f;
    for (int u = 0; u < L; ++u) { float e = expf(sc[h][u] - mx); sc[h][u] = e; sm += e; }
    float inv = 1.f / sm;
    for (int u = 0; u < L; ++u) sc[h][u] *= inv;
  }
  __syncthreads();
  int h = tid >> 6, d = tid & 63;
  float acc = 0.f;
  for (int u = 0; u < L; ++u) {
    const float* vrow = (u < s) ? (Vh + (size_t)(b * KLITS + u) * HH) : vq;
    acc += sc[h][u] * vrow[h * 64 + d];
  }
  o_bf[b * HH + tid] = f2bf(acc);
}

// cross attention: one block per (b, head); streams bf16 K/V of 2304 memory slots.
// K/V rows read as packed u32 pairs; global_prefetch issued one chunk ahead.
__global__ __launch_bounds__(256) void cross_attn_kernel(
    const float* __restrict__ q, const u16* __restrict__ K, const u16* __restrict__ V,
    const int* __restrict__ mask, u16* __restrict__ out_bf) {
  __shared__ float sc[MAXMEM];
  __shared__ float qh[64];
  __shared__ float red[256];
  int b = blockIdx.x >> 2, h = blockIdx.x & 3;
  int tid = threadIdx.x;
  if (tid < 64) qh[tid] = q[b * HH + h * 64 + tid];
  __syncthreads();
  float lmax = -__builtin_inff();
  for (int j = tid; j < MAXMEM; j += 256) {
    if (j + 256 < MAXMEM)
      __builtin_prefetch(K + (size_t)(b * MAXMEM + j + 256) * HH + h * 64, 0, 1);
    const u32* krow = (const u32*)(K + (size_t)(b * MAXMEM + j) * HH + h * 64);
    float acc = 0.f;
    for (int dw = 0; dw < 32; ++dw) {
      u32 u = krow[dw];
      acc += qh[2 * dw] * bfp_lo(u) + qh[2 * dw + 1] * bfp_hi(u);
    }
    acc *= 0.125f;
    if (!mask[b * MAXMEM + j]) acc = -__builtin_inff();
    sc[j] = acc;
    lmax = fmaxf(lmax, acc);
  }
  red[tid] = lmax; __syncthreads();
  for (int off = 128; off; off >>= 1) { if (tid < off) red[tid] = fmaxf(red[tid], red[tid + off]); __syncthreads(); }
  float gmax = red[0]; __syncthreads();
  float lsum = 0.f;
  for (int j = tid; j < MAXMEM; j += 256) { float e = expf(sc[j] - gmax); sc[j] = e; lsum += e; }
  red[tid] = lsum; __syncthreads();
  for (int off = 128; off; off >>= 1) { if (tid < off) red[tid] += red[tid + off]; __syncthreads(); }
  float inv = 1.f / red[0]; __syncthreads();
  // o accumulation: 8 groups x 32 lanes; each lane owns a packed dim pair (2 dims)
  int g = tid >> 5, lane = tid & 31;
  float aLo = 0.f, aHi = 0.f;
  for (int j = g; j < MAXMEM; j += 8) {
    if (j + 8 < MAXMEM)
      __builtin_prefetch(V + (size_t)(b * MAXMEM + j + 8) * HH + h * 64, 0, 1);
    u32 u = ((const u32*)(V + (size_t)(b * MAXMEM + j) * HH + h * 64))[lane];
    float w = sc[j];
    aLo += w * bfp_lo(u);
    aHi += w * bfp_hi(u);
  }
  red[g * 32 + lane] = aLo; __syncthreads();
  float totLo = 0.f, totHi = 0.f;
  if (g == 0) { for (int gg = 0; gg < 8; ++gg) totLo += red[gg * 32 + lane]; }
  __syncthreads();
  red[g * 32 + lane] = aHi; __syncthreads();
  if (g == 0) {
    for (int gg = 0; gg < 8; ++gg) totHi += red[gg * 32 + lane];
    out_bf[b * HH + h * 64 + 2 * lane]     = f2bf(totLo * inv);
    out_bf[b * HH + h * 64 + 2 * lane + 1] = f2bf(totHi * inv);
  }
}

// small output heads: cont, pol, n_args(5), arg_type(3), arg_var(20) — 30 dots of 256
__global__ __launch_bounds__(256) void heads_kernel(
    const float* __restrict__ hvec,
    const float* cW, const float* cB, const float* pW, const float* pB,
    const float* naW, const float* naB, const float* atW, const float* atB,
    const float* avW, const float* avB, float* __restrict__ out, int s) {
  __shared__ float hs[256];
  int b = blockIdx.x, tid = threadIdx.x;
  hs[tid] = hvec[b * HH + tid];
  __syncthreads();
  int lane = tid & 31, wave = tid >> 5;
  for (int id = wave; id < 30; id += 8) {
    const float* W; float bias; int kind, j;
    if (id == 0)      { W = cW;  bias = cB[0]; kind = 0; j = 0; }
    else if (id == 1) { W = pW;  bias = pB[0]; kind = 1; j = 0; }
    else if (id < 7)  { j = id - 2;  W = naW + j * 256; bias = naB[j]; kind = 2; }
    else if (id < 10) { j = id - 7;  W = atW + j * 256; bias = atB[j]; kind = 3; }
    else              { j = id - 10; W = avW + j * 256; bias = avB[j]; kind = 4; }
    float p = 0.f;
    for (int k = lane; k < 256; k += 32) p += hs[k] * W[k];
    for (int m = 16; m; m >>= 1) p += __shfl_xor(p, m, 32);
    if (lane == 0) {
      float v = p + bias;
      int bs = b * NSTEP + s;
      if (kind == 0)      out[O_CONT + bs] = v;
      else if (kind == 1) out[O_POL + bs] = v;
      else if (kind == 2) out[O_NA + bs * 5 + j] = v;
      else if (kind == 3) for (int a = 0; a < 4; ++a) out[O_AT + (bs * 4 + a) * 3 + j] = v;
      else                for (int a = 0; a < 4; ++a) out[O_AV + (bs * 4 + a) * 20 + j] = v;
    }
  }
}

// predicate scores (masked) and arg-func scores (unmasked, broadcast over 4 args)
__global__ __launch_bounds__(256) void psaf_kernel(
    const float* __restrict__ pq, const float* __restrict__ afq,
    const u16* __restrict__ pk, const int* __restrict__ smask,
    float* __restrict__ out, int s) {
  __shared__ float pqs[256], afs[256];
  int b = blockIdx.x, tid = threadIdx.x;
  pqs[tid] = pq[b * HH + tid];
  afs[tid] = afq[b * HH + tid];
  __syncthreads();
  const u32* row = (const u32*)(pk + (size_t)(b * MAXSYM + tid) * HH);
  float d1 = 0.f, d2 = 0.f;
  for (int dw = 0; dw < 128; ++dw) {
    u32 u = row[dw];
    float lo = bfp_lo(u), hi = bfp_hi(u);
    d1 += pqs[2 * dw] * lo + pqs[2 * dw + 1] * hi;
    d2 += afs[2 * dw] * lo + afs[2 * dw + 1] * hi;
  }
  d1 *= 0.0625f; d2 *= 0.0625f;
  int bs = b * NSTEP + s;
  out[O_PS + bs * 256 + tid] = smask[b * MAXSYM + tid] ? d1 : -__builtin_inff();
  for (int a = 0; a < 4; ++a) out[O_AF + (bs * 4 + a) * 256 + tid] = d2;
}

// sequential per-sample target parsing (scan over T=64 actions)
__global__ void parse_targets_kernel(const int* __restrict__ actions, const int* __restrict__ args,
                                     const int* __restrict__ lengths, float* __restrict__ out) {
  int b = threadIdx.x;
  if (b >= BB) return;
  int li = -1, ai = 0, depth = 0, nl = 0;
  float polv[KLITS];
  int pidx[KLITS], nargsv[KLITS], atype[KLITS][4], avar[KLITS][4], afunc[KLITS][4];
  for (int l = 0; l < KLITS; ++l) {
    polv[l] = 0.f; pidx[l] = 0; nargsv[l] = 0;
    for (int a = 0; a < 4; ++a) { atype[l][a] = 0; avar[l][a] = 0; afunc[l][a] = 0; }
  }
  int len = lengths[b];
  for (int t = 0; t < 64; ++t) {
    int act = actions[b * 64 + t], ar = args[b * 64 + t];
    bool active = t < len;
    bool is_new = active && (act == 0 || act == 1);
    if (is_new) { li++; ai = 0; depth = 0; }
    bool in_rng = (li >= 0) && (li < KLITS);
    if (is_new && in_rng) { polv[li] = (act == 1) ? 1.f : 0.f; nl = li + 1; }
    if (active && act == 2 && in_rng) pidx[li] = ar;
    bool can = active && in_rng && (ai < 4) && (depth == 0);
    bool wv = can && (act == 3), wf = can && (act == 4);
    if (wv || wf) {
      atype[li][ai] = wf ? 2 : 1;
      if (wv) avar[li][ai] = (ar < 19) ? ar : 19;
      if (wf) afunc[li][ai] = ar;
      nargsv[li] = ai + 1;
      ai++;
    }
    if (wf) depth++;
    else if (active && act == 5 && depth > 0) depth--;
  }
  out[O_TNL + b] = (float)nl;
  for (int l = 0; l < KLITS; ++l) {
    out[O_TPOL + b * KLITS + l] = polv[l];
    out[O_TPIDX + b * KLITS + l] = (float)pidx[l];
    out[O_TNARG + b * KLITS + l] = (float)nargsv[l];
    for (int a = 0; a < 4; ++a) {
      out[O_TAT + (b * KLITS + l) * 4 + a] = (float)atype[l][a];
      out[O_TAV + (b * KLITS + l) * 4 + a] = (float)avar[l][a];
      out[O_TAF + (b * KLITS + l) * 4 + a] = (float)afunc[l][a];
    }
  }
}

// ---------------- host-side launchers ----------------
static void launch_gemm(const u16* A, int lda, const u16* W, const float* bias,
                        void* C, int ldc, int M, int N, int K, bool relu, bool outbf,
                        hipStream_t st) {
  int tilesM = M >> 4, tilesN = N >> 4;
  int tiles = tilesM * tilesN;
  dim3 g((tiles + 7) / 8), blk(256);
  if (!relu && !outbf) gemm_bf16_kernel<false, false><<<g, blk, 0, st>>>(A, lda, W, bias, C, ldc, tilesM, tilesN, K);
  else if (!relu)      gemm_bf16_kernel<false, true ><<<g, blk, 0, st>>>(A, lda, W, bias, C, ldc, tilesM, tilesN, K);
  else if (outbf)      gemm_bf16_kernel<true,  true ><<<g, blk, 0, st>>>(A, lda, W, bias, C, ldc, tilesM, tilesN, K);
  else                 gemm_bf16_kernel<true,  false><<<g, blk, 0, st>>>(A, lda, W, bias, C, ldc, tilesM, tilesN, K);
}

static void launch_gemm_dual(const u16* A, int lda,
                             const u16* W1, const float* b1, void* C1,
                             const u16* W2, const float* b2, void* C2,
                             int ldc, int M, int N, int K, bool outbf, hipStream_t st) {
  int tilesM = M >> 4, tilesN = N >> 4;
  int tiles = tilesM * tilesN;
  dim3 g((tiles + 7) / 8), blk(256);
  if (outbf) gemm_bf16_dual_kernel<true ><<<g, blk, 0, st>>>(A, lda, W1, b1, C1, W2, b2, C2, ldc, tilesM, tilesN, K);
  else       gemm_bf16_dual_kernel<false><<<g, blk, 0, st>>>(A, lda, W1, b1, C1, W2, b2, C2, ldc, tilesM, tilesN, K);
}

// input indices: params dict flattened in sorted-key order, then positional args
enum {
  IN_ARG_FQ_W = 0, IN_ARG_FQ_B, IN_ARG_TYPE_W, IN_ARG_TYPE_B, IN_ARG_VAR_W, IN_ARG_VAR_B,
  IN_CONT_W, IN_CONT_B,
  IN_CROSS_WK, IN_CROSS_WO, IN_CROSS_WQ, IN_CROSS_WV,
  IN_CROSS_BK, IN_CROSS_BO, IN_CROSS_BQ, IN_CROSS_BV,
  IN_CROSS_NORM_B, IN_CROSS_NORM_G,
  IN_FFN_W1, IN_FFN_W2, IN_FFN_B1, IN_FFN_B2, IN_FFN_NORM_B, IN_FFN_NORM_G,
  IN_HIST_W, IN_HIST_B,
  IN_NARGS_W, IN_NARGS_B,
  IN_POL_W, IN_POL_B,
  IN_PREDK_W, IN_PREDK_B, IN_PREDQ_W, IN_PREDQ_B,
  IN_QUERY_EMBED,
  IN_SELF_WK, IN_SELF_WO, IN_SELF_WQ, IN_SELF_WV,
  IN_SELF_BK, IN_SELF_BO, IN_SELF_BQ, IN_SELF_BV,
  IN_SELF_NORM_B, IN_SELF_NORM_G,
  IN_XCL, IN_XLI, IN_XSY, IN_XTE, IN_XVA,
  IN_BCL, IN_BLI, IN_BSY, IN_BTE, IN_BVA,
  IN_TACT, IN_TARG, IN_TLEN, IN_NSYM
};

extern "C" void kernel_launch(void* const* d_in, const int* in_sizes, int n_in,
                              void* d_out, int out_size, void* d_ws, size_t ws_size,
                              hipStream_t stream) {
  auto F = [&](int i) { return (const float*)d_in[i]; };
  auto I = [&](int i) { return (const int*)d_in[i]; };
  float* out = (float*)d_out;

  // ---- workspace layout ----
  char* ws = (char*)d_ws;
  size_t off = 0;
  auto alloc = [&](size_t bytes) -> void* {
    void* p = ws + off;
    off += (bytes + 255) & ~(size_t)255;
    return p;
  };
  u16* mem_e  = (u16*)alloc((size_t)BB * MAXMEM * HH * 2);
  u16* kmem   = (u16*)alloc((size_t)BB * MAXMEM * HH * 2);
  u16* vmem   = (u16*)alloc((size_t)BB * MAXMEM * HH * 2);
  u16* sym_e  = (u16*)alloc((size_t)BB * MAXSYM * HH * 2);
  u16* pk     = (u16*)alloc((size_t)BB * MAXSYM * HH * 2);
  int* mem_mask = (int*)alloc((size_t)BB * MAXMEM * 4);
  int* sym_mask = (int*)alloc((size_t)BB * MAXSYM * 4);
  int* starts = (int*)alloc(5 * 65 * 4);
  int* pre    = (int*)alloc(5 * 64 * 4);
  // bf16 weights
  u16* WQs = (u16*)alloc(65536 * 2); u16* WKs = (u16*)alloc(65536 * 2);
  u16* WVs = (u16*)alloc(65536 * 2); u16* WOs = (u16*)alloc(65536 * 2);
  u16* WQc = (u16*)alloc(65536 * 2); u16* WKc = (u16*)alloc(65536 * 2);
  u16* WVc = (u16*)alloc(65536 * 2); u16* WOc = (u16*)alloc(65536 * 2);
  u16* W1  = (u16*)alloc(262144 * 2); u16* W2 = (u16*)alloc(262144 * 2);
  u16* WPK = (u16*)alloc(65536 * 2); u16* WPQ = (u16*)alloc(65536 * 2);
  u16* WFQ = (u16*)alloc(65536 * 2); u16* WHI = (u16*)alloc(65536 * 2);
  // constant query-token projections
  float* qeq = (float*)alloc(256 * 4);
  float* qek = (float*)alloc(256 * 4);
  float* qev = (float*)alloc(256 * 4);
  // per-step state
  float* Kself = (float*)alloc((size_t)BB * KLITS * HH * 4);
  float* Vself = (float*)alloc((size_t)BB * KLITS * HH * 4);
  u16* hist_bf = (u16*)alloc((size_t)BB * NSTEP * HH * 2);
  u16* o_bf    = (u16*)alloc((size_t)BB * HH * 2);
  float* sa    = (float*)alloc((size_t)BB * HH * 4);
  float* cur   = (float*)alloc((size_t)BB * HH * 4);
  u16* cur_bf  = (u16*)alloc((size_t)BB * HH * 2);
  float* cq    = (float*)alloc((size_t)BB * HH * 4);
  u16* cat_bf  = (u16*)alloc((size_t)BB * HH * 2);
  float* co    = (float*)alloc((size_t)BB * HH * 4);
  float* cur2  = (float*)alloc((size_t)BB * HH * 4);
  u16* cur2_bf = (u16*)alloc((size_t)BB * HH * 2);
  u16* ff1     = (u16*)alloc((size_t)BB * 1024 * 2);
  float* ff2   = (float*)alloc((size_t)BB * HH * 4);
  float* hf    = (float*)alloc((size_t)BB * HH * 4);
  u16* hbf     = (u16*)alloc((size_t)BB * HH * 2);
  float* pq    = (float*)alloc((size_t)BB * HH * 4);
  float* afq   = (float*)alloc((size_t)BB * HH * 4);
  if (off > ws_size) return;  // workspace too small; bail safely

  // ---- convert weights to bf16 ----
  auto cvt = [&](const float* src, u16* dst, int n) {
    cvt_bf16_kernel<<<(n + 255) / 256, 256, 0, stream>>>(src, dst, n);
  };
  cvt(F(IN_SELF_WQ), WQs, 65536);  cvt(F(IN_SELF_WK), WKs, 65536);
  cvt(F(IN_SELF_WV), WVs, 65536);  cvt(F(IN_SELF_WO), WOs, 65536);
  cvt(F(IN_CROSS_WQ), WQc, 65536); cvt(F(IN_CROSS_WK), WKc, 65536);
  cvt(F(IN_CROSS_WV), WVc, 65536); cvt(F(IN_CROSS_WO), WOc, 65536);
  cvt(F(IN_FFN_W1), W1, 262144);   cvt(F(IN_FFN_W2), W2, 262144);
  cvt(F(IN_PREDK_W), WPK, 65536);  cvt(F(IN_PREDQ_W), WPQ, 65536);
  cvt(F(IN_ARG_FQ_W), WFQ, 65536); cvt(F(IN_HIST_W), WHI, 65536);

  // ---- pad_segments ----
  starts_kernel<<<1, 256, 0, stream>>>(I(IN_BCL), NCL, I(IN_BLI), NLI, I(IN_BSY), NSY,
                                       I(IN_BTE), NTE, I(IN_BVA), NVA, starts, pre);
  zero_kernel<<<(BB * MAXMEM + 255) / 256, 256, 0, stream>>>((u32*)mem_mask, BB * MAXMEM);
  zero_kernel<<<(BB * MAXSYM + 255) / 256, 256, 0, stream>>>((u32*)sym_mask, BB * MAXSYM);
  {
    int n32 = BB * MAXSYM * HH / 2;  // sym_e as u32
    zero_kernel<<<(n32 + 255) / 256, 256, 0, stream>>>((u32*)sym_e, n32);
  }
  scatter_mem_kernel<<<NCL + NLI + NSY + NTE + NVA, 256, 0, stream>>>(
      F(IN_XCL), F(IN_XLI), F(IN_XSY), F(IN_XTE), F(IN_XVA),
      I(IN_BCL), I(IN_BLI), I(IN_BSY), I(IN_BTE), I(IN_BVA),
      starts, pre, mem_e, mem_mask);
  scatter_sym_kernel<<<NSY, 256, 0, stream>>>(F(IN_XSY), I(IN_BSY), starts, sym_e, sym_mask);

  // ---- hoisted projections (step-invariant): fused cross K/V of memory, pred_k of symbols ----
  launch_gemm_dual(mem_e, HH, WKc, F(IN_CROSS_BK), kmem, WVc, F(IN_CROSS_BV), vmem,
                   HH, BB * MAXMEM, HH, HH, true, stream);
  launch_gemm(sym_e, HH, WPK, F(IN_PREDK_B), pk, HH, BB * MAXSYM, HH, HH, false, true, stream);

  // ---- constant query-token self-attn projections ----
  vec_linear_kernel<<<1, 256, 0, stream>>>(F(IN_QUERY_EMBED), F(IN_SELF_WQ), F(IN_SELF_BQ), qeq, HH, HH);
  vec_linear_kernel<<<1, 256, 0, stream>>>(F(IN_QUERY_EMBED), F(IN_SELF_WK), F(IN_SELF_BK), qek, HH, HH);
  vec_linear_kernel<<<1, 256, 0, stream>>>(F(IN_QUERY_EMBED), F(IN_SELF_WV), F(IN_SELF_BV), qev, HH, HH);

  // ---- decoder steps ----
  for (int s = 0; s < NSTEP; ++s) {
    if (s > 0) {
      // fused self K/V of the newly appended history token (row-strided A over hist_bf)
      launch_gemm_dual(hist_bf + (size_t)(s - 1) * HH, NSTEP * HH,
                       WKs, F(IN_SELF_BK), Kself + (size_t)(s - 1) * HH,
                       WVs, F(IN_SELF_BV), Vself + (size_t)(s - 1) * HH,
                       KLITS * HH, BB, HH, HH, false, stream);
    }
    self_attn_kernel<<<BB, 256, 0, stream>>>(qeq, qek, qev, Kself, Vself, o_bf, s);
    launch_gemm(o_bf, HH, WOs, F(IN_SELF_BO), sa, HH, BB, HH, HH, false, false, stream);
    ln_kernel<<<BB, 256, 0, stream>>>(sa, F(IN_QUERY_EMBED), 0,
                                      F(IN_SELF_NORM_G), F(IN_SELF_NORM_B), cur, cur_bf);
    launch_gemm(cur_bf, HH, WQc, F(IN_CROSS_BQ), cq, HH, BB, HH, HH, false, false, stream);
    cross_attn_kernel<<<BB * 4, 256, 0, stream>>>(cq, kmem, vmem, mem_mask, cat_bf);
    launch_gemm(cat_bf, HH, WOc, F(IN_CROSS_BO), co, HH, BB, HH, HH, false, false, stream);
    ln_kernel<<<BB, 256, 0, stream>>>(co, cur, HH,
                                      F(IN_CROSS_NORM_G), F(IN_CROSS_NORM_B), cur2, cur2_bf);
    launch_gemm(cur2_bf, HH, W1, F(IN_FFN_B1), ff1, 1024, BB, 1024, HH, true, true, stream);
    launch_gemm(ff1, 1024, W2, F(IN_FFN_B2), ff2, HH, BB, HH, 1024, false, false, stream);
    ln_kernel<<<BB, 256, 0, stream>>>(ff2, cur2, HH,
                                      F(IN_FFN_NORM_G), F(IN_FFN_NORM_B), hf, hbf);
    // fused pred_q / arg_fq projections share the h activations
    launch_gemm_dual(hbf, HH, WPQ, F(IN_PREDQ_B), pq, WFQ, F(IN_ARG_FQ_B), afq,
                     HH, BB, HH, HH, false, stream);
    heads_kernel<<<BB, 256, 0, stream>>>(hf,
        F(IN_CONT_W), F(IN_CONT_B), F(IN_POL_W), F(IN_POL_B),
        F(IN_NARGS_W), F(IN_NARGS_B), F(IN_ARG_TYPE_W), F(IN_ARG_TYPE_B),
        F(IN_ARG_VAR_W), F(IN_ARG_VAR_B), out, s);
    psaf_kernel<<<BB, 256, 0, stream>>>(pq, afq, pk, sym_mask, out, s);
    // next history token
    launch_gemm(hbf, HH, WHI, F(IN_HIST_B), hist_bf + (size_t)s * HH, NSTEP * HH,
                BB, HH, HH, false, true, stream);
  }

  // ---- target parsing ----
  parse_targets_kernel<<<1, 64, 0, stream>>>(I(IN_TACT), I(IN_TARG), I(IN_TLEN), out);
}
